// TrajectoryGenerator_88845693485341
// MI455X (gfx1250) — matrix operand order repaired and checked by reading, not compiled, not run on hardware
//
#include <hip/hip_runtime.h>
#include <hip/hip_bf16.h>
#include <math.h>

// ---------------------------------------------------------------------------
// CDNA5 / gfx1250 trajectory LSTM enc-dec, spatial + temporal attention.
// GEMMs: v_wmma_f32_16x16x32_f16 (wave32, 16x16 tile per wave, K=64).
// h-block staging: global_load_async_to_lds_b128 (+ s_wait_asynccnt).
// Activations: hardware transcendentals (v_tanh_f32 / v_exp_f32 / v_rcp_f32).
// ---------------------------------------------------------------------------

#define USE_ASYNC_LDS 1

typedef __attribute__((ext_vector_type(16))) _Float16 v16h;
typedef __attribute__((ext_vector_type(8)))  float    v8f;
typedef __attribute__((address_space(3))) char lds_char_t;

constexpr int BB    = 512;
constexpr int PP    = 64;
constexpr int TOBS  = 8;
constexpr int TPRED = 12;
constexpr int TP    = BB * PP;       // 32768
constexpr int HD    = 32;
constexpr int GDIM  = 128;           // 4*HD gates
constexpr float INV_SCALE = 0.17677669529663687f;   // 1/sqrt(32)
constexpr float DEG   = 57.29577951308232f;
constexpr float L2E   = 1.4426950408889634f;        // log2(e)
constexpr float L2E2  = 2.8853900817779268f;        // 2*log2(e)

// ---- hardware transcendental helpers (TRANS pipe, no libm branches) -------
static __device__ __forceinline__ float fast_exp_(float x)
{
    return __builtin_amdgcn_exp2f(x * L2E);                    // v_exp_f32
}
static __device__ __forceinline__ float fast_tanh_(float x)
{
#if __has_builtin(__builtin_amdgcn_tanhf)
    return __builtin_amdgcn_tanhf(x);                          // v_tanh_f32
#elif __has_builtin(__builtin_amdgcn_tanh_f32)
    return __builtin_amdgcn_tanh_f32(x);
#else
    // tanh(x) = 1 - 2/(2^(2x*log2e) + 1)
    float e = __builtin_amdgcn_exp2f(x * L2E2);
    return 1.0f - 2.0f * __builtin_amdgcn_rcpf(e + 1.0f);
#endif
}
static __device__ __forceinline__ float sigm_(float x)
{
    return __builtin_amdgcn_rcpf(1.0f + __builtin_amdgcn_exp2f(-x * L2E));
}
static __device__ __forceinline__ float mod360_(float x)
{
    return x - floorf(x * (1.0f / 360.0f)) * 360.0f;
}

// ---------------------------------------------------------------------------
// One-time weight packing: f32 -> f16 B-panels, K-major (64 x N), fused biases
// ---------------------------------------------------------------------------
__global__ void pack_weights_k(const float* __restrict__ Wih_e, const float* __restrict__ Whh_e,
                               const float* __restrict__ bih_e, const float* __restrict__ bhh_e,
                               const float* __restrict__ Wih_d, const float* __restrict__ Whh_d,
                               const float* __restrict__ bih_d, const float* __restrict__ bhh_d,
                               const float* __restrict__ Wtemp,
                               _Float16* __restrict__ We16, _Float16* __restrict__ Wd16,
                               _Float16* __restrict__ Wt16,
                               float* __restrict__ bias_e, float* __restrict__ bias_d)
{
    int idx = blockIdx.x * blockDim.x + threadIdx.x;          // 0 .. 8191
    if (idx < 64 * GDIM) {
        int k = idx / GDIM, n = idx % GDIM;
        float we = (k < 16) ? Wih_e[n * 16 + k] : (k < 48 ? Whh_e[n * 32 + (k - 16)] : 0.0f);
        float wd = (k < 16) ? Wih_d[n * 16 + k] : (k < 48 ? Whh_d[n * 32 + (k - 16)] : 0.0f);
        We16[k * GDIM + n] = (_Float16)we;
        Wd16[k * GDIM + n] = (_Float16)wd;
    }
    if (idx < 64 * HD) {                                      // W_temp (32,64) -> (64,32) K-major
        int k = idx / HD, n = idx % HD;
        Wt16[k * HD + n] = (_Float16)Wtemp[n * 64 + k];
    }
    if (idx < GDIM) {
        bias_e[idx] = bih_e[idx] + bhh_e[idx];
        bias_d[idx] = bih_d[idx] + bhh_d[idx];
    }
}

__global__ void zero_f32_k(float* __restrict__ p, int n)
{
    int idx = blockIdx.x * blockDim.x + threadIdx.x;
    if (idx < n) p[idx] = 0.0f;
}

// ---------------------------------------------------------------------------
// Generic WMMA GEMM:  C(M,N) = act( A16(M,64) @ B16(64,N) + bias(N) )
// one 16x16 tile per wave; 8 waves / 256-thread block; K = 64 = 2 slices of 32
// ---------------------------------------------------------------------------
__global__ void wmma_gemm_a64_k(const _Float16* __restrict__ A,
                                const _Float16* __restrict__ Bw,
                                const float* __restrict__ bias,
                                float* __restrict__ C,
                                int N, int ntilesN, int totalTiles, int act)
{
    int wave = threadIdx.x >> 5;
    int lane = threadIdx.x & 31;
    int tile = blockIdx.x * 8 + wave;
    if (tile >= totalTiles) return;                            // wave-uniform
    int mtile = tile / ntilesN;
    int ntile = tile - mtile * ntilesN;
    int row0  = mtile * 16;
    int n0    = ntile * 16;

    // A fragment (ISA 7.12.2, 16-bit A 16x32): lane -> M, halves -> K groups
    int mrow  = lane & 15;
    int kbase = (lane < 16) ? 0 : 8;
    const _Float16* arow = A + (size_t)(row0 + mrow) * 64;
    __builtin_prefetch(arow + 64 * 16, 0, 1);                  // global_prefetch_b8

    v8f acc = {};
#pragma unroll
    for (int s = 0; s < 2; ++s) {
        v16h af, bf;
        const _Float16* ap = arow + s * 32 + kbase;
#pragma unroll
        for (int e = 0; e < 8; ++e) { af[e] = ap[e]; af[8 + e] = ap[16 + e]; }
        // B fragment (K x N): lane -> N, halves -> K groups (same as A)
        const _Float16* bp = Bw + (size_t)(s * 32 + kbase) * N + n0 + mrow;
#pragma unroll
        for (int e = 0; e < 8; ++e) { bf[e] = bp[(size_t)e * N]; bf[8 + e] = bp[(size_t)(16 + e) * N]; }
        acc = __builtin_amdgcn_wmma_f32_16x16x32_f16(false, af, false, bf,
                                                     (short)0, acc, false, false);
    }

    // D layout: lane -> N, vgpr v -> M = v + (lane>=16 ? 8 : 0)
    int ncol  = lane & 15;
    int mbase = (lane < 16) ? 0 : 8;
    float bv = bias ? bias[n0 + ncol] : 0.0f;
#pragma unroll
    for (int v = 0; v < 8; ++v) {
        float val = acc[v] + bv;
        if (act) val = fast_tanh_(val);
        C[(size_t)(row0 + mbase + v) * N + n0 + ncol] = val;
    }
}

// ---------------------------------------------------------------------------
// Encoder prep: A16 row = [ x_t @ Wemb^T + b | h_prev | zeros ]  (f16, 64 wide)
// ---------------------------------------------------------------------------
__global__ void enc_prep_k(const float* __restrict__ x, int t,
                           const float* __restrict__ Wemb, const float* __restrict__ bemb,
                           const float* __restrict__ hprev, _Float16* __restrict__ A16)
{
    int r = blockIdx.x * blockDim.x + threadIdx.x;
    if (r >= TP) return;
    float x0 = x[((size_t)r * TOBS + t) * 2 + 0];
    float x1 = x[((size_t)r * TOBS + t) * 2 + 1];
    _Float16* row = A16 + (size_t)r * 64;
#pragma unroll
    for (int j = 0; j < 16; ++j)
        row[j] = (_Float16)(x0 * Wemb[j * 2] + x1 * Wemb[j * 2 + 1] + bemb[j]);
    const float* hr = hprev + (size_t)r * HD;
#pragma unroll
    for (int k = 0; k < HD; ++k) row[16 + k] = (_Float16)hr[k];
#pragma unroll
    for (int k = 48; k < 64; ++k) row[k] = (_Float16)0.0f;
}

// Decoder prep: embedding (cols 0..15) + zero pad (48..63) of A16.
// Cols 16..47 are written by spatial_attn_k mode 1.
__global__ void dec_prep_k(const float* __restrict__ prev_x,
                           const float* __restrict__ Wemb, const float* __restrict__ bemb,
                           _Float16* __restrict__ A16)
{
    int r = blockIdx.x * blockDim.x + threadIdx.x;
    if (r >= TP) return;
    float x0 = prev_x[(size_t)r * 2 + 0];
    float x1 = prev_x[(size_t)r * 2 + 1];
    _Float16* row = A16 + (size_t)r * 64;
#pragma unroll
    for (int j = 0; j < 16; ++j)
        row[j] = (_Float16)(x0 * Wemb[j * 2] + x1 * Wemb[j * 2 + 1] + bemb[j]);
#pragma unroll
    for (int k = 48; k < 64; ++k) row[k] = (_Float16)0.0f;
}

// ---------------------------------------------------------------------------
// LSTM gate elementwise: g(TP,128) -> c, h  (i,f,g,o split)
// ---------------------------------------------------------------------------
__global__ void lstm_gates_k(const float* __restrict__ g,
                             float* __restrict__ c, float* __restrict__ hout)
{
    int idx = blockIdx.x * blockDim.x + threadIdx.x;           // TP*32
    if (idx >= TP * HD) return;
    int r = idx >> 5, j = idx & 31;
    const float* gr = g + (size_t)r * GDIM;
    float i_ = sigm_(gr[j]);
    float f_ = sigm_(gr[32 + j]);
    float gg = fast_tanh_(gr[64 + j]);
    float o_ = sigm_(gr[96 + j]);
    float cn = f_ * c[idx] + i_ * gg;
    c[idx]    = cn;
    hout[idx] = o_ * fast_tanh_(cn);
}

// ---------------------------------------------------------------------------
// Spatial attention, one workgroup per batch (256 thr = 8 waves):
//   async-DMA the 64x32 f32 h-block into LDS, build w(64,64), row-normalize,
//   f16-convert, 8 WMMA tiles of tanh( w @ h ).
//   mode 0: write f32 h-carry + encoded[t];  mode 1: f16 into A16 cols 16..47
// ---------------------------------------------------------------------------
__global__ void spatial_attn_k(const float* __restrict__ hin,
                               const float* __restrict__ dmat,
                               const float* __restrict__ bmat,
                               const float* __restrict__ hmatm,
                               const float* __restrict__ maskp, int mstride, int moff,
                               const float* __restrict__ domain,
                               int rstride, int coff,
                               int mode,
                               float* __restrict__ hout,
                               float* __restrict__ encout, int enc_t,
                               _Float16* __restrict__ a16out)
{
    __shared__ float    hstage[64 * 32];      // async landing zone (f32)
    __shared__ float    wf[64 * 64];
    __shared__ _Float16 w16[64 * 64];
    __shared__ _Float16 hb16[64 * 32];
    __shared__ float    psum[64 * 4];
    __shared__ float    rinv[64];
    __shared__ float    msk[64];

    int b   = blockIdx.x;
    int tid = threadIdx.x;

    const float* hsrc = hin + (size_t)(b * 64) * HD;           // 8KB block

#if USE_ASYNC_LDS
    // Each thread DMAs 2 x 16B directly into LDS (ASYNCcnt-tracked).
    {
        unsigned lbase = (unsigned)(unsigned long long)(lds_char_t*)&hstage[0];
        unsigned l0 = lbase + (unsigned)tid * 16u;
        unsigned l1 = l0 + 4096u;
        unsigned long long g0 = (unsigned long long)(const char*)hsrc + (unsigned)tid * 16u;
        unsigned long long g1 = g0 + 4096u;
        asm volatile("global_load_async_to_lds_b128 %0, %1, off"
                     :: "v"(l0), "v"(g0) : "memory");
        asm volatile("global_load_async_to_lds_b128 %0, %1, off"
                     :: "v"(l1), "v"(g1) : "memory");
        asm volatile("s_wait_asynccnt 0x0" ::: "memory");
    }
#else
#pragma unroll
    for (int e = 0; e < 8; ++e) {
        int idx = tid * 8 + e;
        hstage[idx] = hsrc[idx];
    }
#endif

    if (tid < 64) msk[tid] = maskp[(size_t)(b * 64 + tid) * mstride + moff];
    __syncthreads();

    // f32 -> f16 conversion of the h block (2048 elems / 256 thr)
#pragma unroll
    for (int e = 0; e < 8; ++e) {
        int idx = tid * 8 + e;
        hb16[idx] = (_Float16)hstage[idx];
    }

    int i = tid >> 2, q = tid & 3;
    float mi = msk[i];
    size_t rowbase = (size_t)(b * 64 + i) * rstride + coff;
    float s = 0.0f;
#pragma unroll
    for (int jj = 0; jj < 16; ++jj) {
        int j = q * 16 + jj;
        float d  = dmat[rowbase + j];
        float be = bmat[rowbase + j];
        float he = hmatm[rowbase + j];
        int ib = min(max((int)floorf(be * (1.0f / 30.0f)), 0), 11);
        int ih = min(max((int)floorf(he * (1.0f / 30.0f)), 0), 11);
        float wv = fmaxf(domain[ih * 12 + ib] - d, 0.0f) * mi * msk[j];
        wf[i * 64 + j] = wv;
        s += wv;
    }
    psum[i * 4 + q] = s;
    __syncthreads();
    if (tid < 64)
        rinv[tid] = __builtin_amdgcn_rcpf(psum[tid * 4] + psum[tid * 4 + 1] +
                                          psum[tid * 4 + 2] + psum[tid * 4 + 3] + 1e-8f);
    __syncthreads();
    float ri = rinv[i];
#pragma unroll
    for (int jj = 0; jj < 16; ++jj) {
        int j = q * 16 + jj;
        w16[i * 64 + j] = (_Float16)(wf[i * 64 + j] * ri);
    }
    __syncthreads();

    // 8 waves -> 4 m-tiles x 2 n-tiles of the 64x64 @ 64x32 product
    int wave = tid >> 5, lane = tid & 31;
    int mtile = wave >> 1, ntile = wave & 1;
    int mrow  = lane & 15;
    int kbase = (lane < 16) ? 0 : 8;
    v8f acc = {};
#pragma unroll
    for (int s2 = 0; s2 < 2; ++s2) {
        v16h af, bf;
        const _Float16* ap = &w16[(mtile * 16 + mrow) * 64 + s2 * 32 + kbase];
#pragma unroll
        for (int e = 0; e < 8; ++e) { af[e] = ap[e]; af[8 + e] = ap[16 + e]; }
        const _Float16* bp = &hb16[(s2 * 32 + kbase) * HD + ntile * 16 + mrow];
#pragma unroll
        for (int e = 0; e < 8; ++e) { bf[e] = bp[e * HD]; bf[8 + e] = bp[(16 + e) * HD]; }
        acc = __builtin_amdgcn_wmma_f32_16x16x32_f16(false, af, false, bf,
                                                     (short)0, acc, false, false);
    }

    int ncol  = lane & 15;
    int mbase = (lane < 16) ? 0 : 8;
#pragma unroll
    for (int v = 0; v < 8; ++v) {
        int ii = mtile * 16 + mbase + v;
        int nn = ntile * 16 + ncol;
        float val = fast_tanh_(acc[v]);
        if (mode == 0) {
            hout[(size_t)(b * 64 + ii) * HD + nn] = val;
            encout[((size_t)(b * 64 + ii) * TOBS + enc_t) * HD + nn] = val;
        } else {
            a16out[(size_t)(b * 64 + ii) * 64 + 16 + nn] = (_Float16)val;
        }
    }
}

// ---------------------------------------------------------------------------
// Temporal attention: per row r, scores over T_OBS, softmax, ctx; writes
// A2 row = [ctx(32) | hc(32)] as f16 for the W_temp WMMA GEMM.
// ---------------------------------------------------------------------------
__global__ void temporal_attn_k(const float* __restrict__ hc,
                                const float* __restrict__ enc,
                                const float* __restrict__ imask,
                                _Float16* __restrict__ A2)
{
    int r = blockIdx.x * blockDim.x + threadIdx.x;
    if (r >= TP) return;
    float hv[HD];
#pragma unroll
    for (int k = 0; k < HD; ++k) hv[k] = hc[(size_t)r * HD + k];

    float sc[TOBS];
    float mx = -1e30f;
#pragma unroll
    for (int t = 0; t < TOBS; ++t) {
        const float* er = enc + ((size_t)r * TOBS + t) * HD;
        float s = 0.0f;
#pragma unroll
        for (int k = 0; k < HD; ++k) s += hv[k] * er[k];
        s *= INV_SCALE;
        sc[t] = (imask[(size_t)r * TOBS + t] > 0.0f) ? s : -1.0e9f;
        mx = fmaxf(mx, sc[t]);
    }
    float den = 0.0f;
#pragma unroll
    for (int t = 0; t < TOBS; ++t) { sc[t] = fast_exp_(sc[t] - mx); den += sc[t]; }
    float inv = __builtin_amdgcn_rcpf(den);

    float ctx[HD];
#pragma unroll
    for (int k = 0; k < HD; ++k) ctx[k] = 0.0f;
#pragma unroll
    for (int t = 0; t < TOBS; ++t) {
        float a = sc[t] * inv;
        const float* er = enc + ((size_t)r * TOBS + t) * HD;
#pragma unroll
        for (int k = 0; k < HD; ++k) ctx[k] += a * er[k];
    }

    _Float16* row = A2 + (size_t)r * 64;
#pragma unroll
    for (int k = 0; k < HD; ++k) row[k]      = (_Float16)ctx[k];
#pragma unroll
    for (int k = 0; k < HD; ++k) row[32 + k] = (_Float16)hv[k];
}

// ---------------------------------------------------------------------------
// Decoder loop-carried geometry init (slice t = T_OBS-1 of inputs)
// ---------------------------------------------------------------------------
__global__ void dec_init_k(const float* __restrict__ x,
                           const float* __restrict__ dmat, const float* __restrict__ bmat,
                           const float* __restrict__ hmat,
                           float* __restrict__ prev_x,
                           float* __restrict__ pd, float* __restrict__ pb, float* __restrict__ ph)
{
    int idx = blockIdx.x * blockDim.x + threadIdx.x;           // TP*64
    if (idx >= TP * PP) return;
    int r = idx / PP, j = idx % PP;
    size_t src = ((size_t)r * TOBS + (TOBS - 1)) * PP + j;
    pd[idx] = dmat[src];
    pb[idx] = bmat[src];
    ph[idx] = hmat[src];
    if (j < 2)
        prev_x[(size_t)r * 2 + j] = x[((size_t)r * TOBS + (TOBS - 1)) * 2 + j];
}

// ---------------------------------------------------------------------------
// Output head + geometry features: one workgroup per batch.
// ---------------------------------------------------------------------------
__global__ void out_feat_k(const float* __restrict__ h,
                           const float* __restrict__ Wout, const float* __restrict__ bout,
                           const float* __restrict__ mean, const float* __restrict__ var,
                           float* __restrict__ prev_x,
                           float* __restrict__ pd, float* __restrict__ pb, float* __restrict__ ph,
                           float* __restrict__ out, int step)
{
    __shared__ float posx[64], posy[64], head[64];
    int b = blockIdx.x, tid = threadIdx.x;

    if (tid < 64) {
        int r = b * 64 + tid;
        const float* hr = h + (size_t)r * HD;
        float s0 = bout[0], s1 = bout[1];
#pragma unroll
        for (int k = 0; k < HD; ++k) { s0 += hr[k] * Wout[k]; s1 += hr[k] * Wout[32 + k]; }
        float xo0 = fast_tanh_(s0), xo1 = fast_tanh_(s1);
        float vx = var[b * 2 + 0], vy = var[b * 2 + 1];
        float mx = mean[b * 2 + 0], my = mean[b * 2 + 1];
        float po0 = prev_x[(size_t)r * 2 + 0], po1 = prev_x[(size_t)r * 2 + 1];
        float p0 = xo0 * vx + mx, p1 = xo1 * vy + my;
        float q0 = po0 * vx + mx, q1 = po1 * vy + my;
        float hd = mod360_(atan2f(p1 - q1, p0 - q0) * DEG);
        posx[tid] = p0; posy[tid] = p1; head[tid] = hd;
        prev_x[(size_t)r * 2 + 0] = xo0;
        prev_x[(size_t)r * 2 + 1] = xo1;
        out[((size_t)r * TPRED + step) * 2 + 0] = xo0;
        out[((size_t)r * TPRED + step) * 2 + 1] = xo1;
    }
    __syncthreads();

    int i = tid >> 2, q = tid & 3;
    float hi = head[i], pix = posx[i], piy = posy[i];
#pragma unroll
    for (int jj = 0; jj < 16; ++jj) {
        int j = q * 16 + jj;
        float rx = posx[j] - pix, ry = posy[j] - piy;
        float dist = sqrtf(rx * rx + ry * ry + 1e-12f);
        float bear = mod360_(atan2f(ry, rx) * DEG);
        size_t o = (size_t)(b * 64 + i) * PP + j;
        pd[o] = dist;
        pb[o] = mod360_(bear - hi);
        ph[o] = mod360_(head[j] - hi);
    }
}

// ---------------------------------------------------------------------------
// Host launch
// ---------------------------------------------------------------------------
extern "C" void kernel_launch(void* const* d_in, const int* in_sizes, int n_in,
                              void* d_out, int out_size, void* d_ws, size_t ws_size,
                              hipStream_t stream)
{
    const float* x      = (const float*)d_in[0];
    const float* dmat   = (const float*)d_in[1];
    const float* bmat   = (const float*)d_in[2];
    const float* hmat   = (const float*)d_in[3];
    const float* imask  = (const float*)d_in[4];
    const float* bmean  = (const float*)d_in[7];
    const float* bvar   = (const float*)d_in[8];
    const float* Wemb   = (const float*)d_in[9];
    const float* bemb   = (const float*)d_in[10];
    const float* Wih_e  = (const float*)d_in[11];
    const float* Whh_e  = (const float*)d_in[12];
    const float* bih_e  = (const float*)d_in[13];
    const float* bhh_e  = (const float*)d_in[14];
    const float* Wih_d  = (const float*)d_in[15];
    const float* Whh_d  = (const float*)d_in[16];
    const float* bih_d  = (const float*)d_in[17];
    const float* bhh_d  = (const float*)d_in[18];
    const float* domain = (const float*)d_in[19];
    const float* Wtemp  = (const float*)d_in[20];
    const float* btemp  = (const float*)d_in[21];
    const float* Wout   = (const float*)d_in[22];
    const float* bout   = (const float*)d_in[23];
    float* out = (float*)d_out;

    // ---- workspace layout (256B aligned) ----
    char* base = (char*)d_ws;
    size_t off = 0;
    auto alloc = [&](size_t bytes) { size_t o = off; off = (off + bytes + 255) & ~(size_t)255; return o; };
    size_t oA16   = alloc((size_t)TP * 64 * sizeof(_Float16));
    size_t oA2    = alloc((size_t)TP * 64 * sizeof(_Float16));
    size_t oG     = alloc((size_t)TP * GDIM * sizeof(float));
    size_t oH     = alloc((size_t)TP * HD * sizeof(float));
    size_t oC     = alloc((size_t)TP * HD * sizeof(float));
    size_t oHC    = alloc((size_t)TP * HD * sizeof(float));
    size_t oENC   = alloc((size_t)TP * TOBS * HD * sizeof(float));
    size_t oPD    = alloc((size_t)TP * PP * sizeof(float));
    size_t oPB    = alloc((size_t)TP * PP * sizeof(float));
    size_t oPH    = alloc((size_t)TP * PP * sizeof(float));
    size_t oPX    = alloc((size_t)TP * 2 * sizeof(float));
    size_t oWE16  = alloc((size_t)64 * GDIM * sizeof(_Float16));
    size_t oWD16  = alloc((size_t)64 * GDIM * sizeof(_Float16));
    size_t oWT16  = alloc((size_t)64 * HD * sizeof(_Float16));
    size_t oBE    = alloc((size_t)GDIM * sizeof(float));
    size_t oBD    = alloc((size_t)GDIM * sizeof(float));
    (void)ws_size; (void)in_sizes; (void)n_in; (void)out_size;

    _Float16* A16  = (_Float16*)(base + oA16);
    _Float16* A2   = (_Float16*)(base + oA2);
    float*    G    = (float*)(base + oG);
    float*    H    = (float*)(base + oH);
    float*    C    = (float*)(base + oC);
    float*    HC   = (float*)(base + oHC);
    float*    ENC  = (float*)(base + oENC);
    float*    PD   = (float*)(base + oPD);
    float*    PB   = (float*)(base + oPB);
    float*    PH   = (float*)(base + oPH);
    float*    PX   = (float*)(base + oPX);
    _Float16* WE16 = (_Float16*)(base + oWE16);
    _Float16* WD16 = (_Float16*)(base + oWD16);
    _Float16* WT16 = (_Float16*)(base + oWT16);
    float*    BE   = (float*)(base + oBE);
    float*    BD   = (float*)(base + oBD);

    const int TB = 256;
    pack_weights_k<<<(64 * GDIM + TB - 1) / TB, TB, 0, stream>>>(
        Wih_e, Whh_e, bih_e, bhh_e, Wih_d, Whh_d, bih_d, bhh_d, Wtemp,
        WE16, WD16, WT16, BE, BD);
    zero_f32_k<<<(TP * HD + TB - 1) / TB, TB, 0, stream>>>(H, TP * HD);
    zero_f32_k<<<(TP * HD + TB - 1) / TB, TB, 0, stream>>>(C, TP * HD);

    const int gatesTiles = (TP / 16) * (GDIM / 16);    // 16384
    const int tempTiles  = (TP / 16) * (HD / 16);      // 4096

    // ---------------- encoder ----------------
    for (int t = 0; t < TOBS; ++t) {
        enc_prep_k<<<TP / TB, TB, 0, stream>>>(x, t, Wemb, bemb, H, A16);
        wmma_gemm_a64_k<<<gatesTiles / 8, TB, 0, stream>>>(A16, WE16, BE, G,
                                                           GDIM, GDIM / 16, gatesTiles, 0);
        lstm_gates_k<<<(TP * HD) / TB, TB, 0, stream>>>(G, C, HC);
        spatial_attn_k<<<BB, TB, 0, stream>>>(HC, dmat, bmat, hmat,
                                              imask, TOBS, t, domain,
                                              TOBS * PP, t * PP,
                                              0, H, ENC, t, (_Float16*)nullptr);
    }

    // ---------------- decoder init ----------------
    dec_init_k<<<(TP * PP) / TB, TB, 0, stream>>>(x, dmat, bmat, hmat, PX, PD, PB, PH);
    zero_f32_k<<<(TP * HD + TB - 1) / TB, TB, 0, stream>>>(C, TP * HD);   // decoder c0 = 0

    // ---------------- decoder ----------------
    for (int s = 0; s < TPRED; ++s) {
        dec_prep_k<<<TP / TB, TB, 0, stream>>>(PX, Wemb, bemb, A16);
        spatial_attn_k<<<BB, TB, 0, stream>>>(H, PD, PB, PH,
                                              imask, TOBS, TOBS - 1, domain,
                                              PP, 0,
                                              1, (float*)nullptr, (float*)nullptr, 0, A16);
        wmma_gemm_a64_k<<<gatesTiles / 8, TB, 0, stream>>>(A16, WD16, BD, G,
                                                           GDIM, GDIM / 16, gatesTiles, 0);
        lstm_gates_k<<<(TP * HD) / TB, TB, 0, stream>>>(G, C, HC);
        temporal_attn_k<<<TP / TB, TB, 0, stream>>>(HC, ENC, imask, A2);
        wmma_gemm_a64_k<<<tempTiles / 8, TB, 0, stream>>>(A2, WT16, btemp, H,
                                                          HD, HD / 16, tempTiles, 1);
        out_feat_k<<<BB, TB, 0, stream>>>(H, Wout, bout, bmean, bvar,
                                          PX, PD, PB, PH, out, s);
    }
}
